// MultiHyenaOperator_51513837748502
// MI455X (gfx1250) — matrix operand order repaired and checked by reading, not compiled, not run on hardware
//
#include <hip/hip_runtime.h>
#include <stdint.h>

#define D_MODEL   768
#define NUM_HEADS 48
#define HEAD_CH   16
#define INNER     2304
#define BATCH     2
#define SEQLEN    2048
#define SKIPV     1.0f
#define EPSV      1e-6f

typedef __attribute__((ext_vector_type(16))) __bf16 v16bf;
typedef __attribute__((ext_vector_type(8)))  float  v8f;

union Frag {
  v16bf v;
  uint32_t u[8];
  uint4 q[2];
};
static_assert(sizeof(Frag) == 32, "frag size");

__device__ __forceinline__ uint16_t f2bf(float f) {
  uint32_t x = __builtin_bit_cast(uint32_t, f);
  uint32_t r = (x + 0x7FFFu + ((x >> 16) & 1u)) >> 16;   // round-to-nearest-even
  return (uint16_t)r;
}

__device__ __forceinline__ v8f wmma_bf16(Frag a, Frag b, v8f c) {
  // D = A(16x32 bf16) x B(32x16 bf16) + C(16x16 f32)
  return __builtin_amdgcn_wmma_f32_16x16x32_bf16(false, a.v, false, b.v,
                                                 (short)0, c, false, false);
}

// A-operand (16xK rows, bf16 row-major): lanes0-15: K = k0+0..7 & k0+16..23,
// lanes16-31: K = k0+8..15 & k0+24..31  -> two b128 at elem (k0+lanehi*8), +16
__device__ __forceinline__ void loadA(Frag& f, const uint16_t* row, int k0, int lanehi) {
  const uint16_t* p = row + k0 + lanehi * 8;
  f.q[0] = *reinterpret_cast<const uint4*>(p);
  f.q[1] = *reinterpret_cast<const uint4*>(p + 16);
}
// B-operand (Kx16, source stored as 16 rows x K contiguous): lanes0-15: K=k0+0..7 & k0+16..23,
// lanes16-31: K=k0+8..15 & k0+24..31 -> two b128 at elem (k0+lanehi*8), +16
__device__ __forceinline__ void loadB(Frag& f, const uint16_t* row, int k0, int lanehi) {
  const uint16_t* p = row + k0 + lanehi * 8;
  f.q[0] = *reinterpret_cast<const uint4*>(p);
  f.q[1] = *reinterpret_cast<const uint4*>(p + 16);
}

// ---------------- f32 -> bf16 convert ----------------
__global__ void __launch_bounds__(256)
k_cvt_bf16(const float* __restrict__ src, uint16_t* __restrict__ dst, int n) {
  int i = blockIdx.x * blockDim.x + threadIdx.x;
  if (i < n) dst[i] = f2bf(src[i]);
}

// ---------------- in-projection GEMM: u[b,e,l] = inputs[b,l,:]·w_in[e,:] + b_in[e]
// block = 256 thr (8 waves). wave: 32e x 64l, rolled ping-pong K pipeline. grid (8, 36, B)
__global__ void __launch_bounds__(256)
k_in_gemm(const uint16_t* __restrict__ w_in_bf,
          const uint16_t* __restrict__ in_bf,
          const float*    __restrict__ b_in,
          float*          __restrict__ u_buf) {
  const int lane = threadIdx.x & 31, wave = threadIdx.x >> 5;
  const int lanehi = lane >> 4, lanelo = lane & 15;
  const int b  = blockIdx.z;
  const int e0 = blockIdx.y * 64 + (wave & 1) * 32;
  const int l0 = blockIdx.x * 256 + (wave >> 1) * 64;

  const uint16_t* arow0 = w_in_bf + (size_t)(e0 + lanelo) * D_MODEL;
  const uint16_t* arow1 = arow0 + (size_t)16 * D_MODEL;
  const uint16_t* brow  = in_bf + ((size_t)b * SEQLEN + l0 + lanelo) * D_MODEL;

  v8f acc[2][4];
  const v8f z8 = {0.f,0.f,0.f,0.f,0.f,0.f,0.f,0.f};
#pragma unroll
  for (int i = 0; i < 2; ++i)
#pragma unroll
    for (int t = 0; t < 4; ++t) acc[i][t] = z8;

  Frag a0[2], b0[4], a1[2], b1[4];
  loadA(a0[0], arow0, 0, lanehi);
  loadA(a0[1], arow1, 0, lanehi);
#pragma unroll
  for (int t = 0; t < 4; ++t) loadB(b0[t], brow + (size_t)t * 16 * D_MODEL, 0, lanehi);

  // keep rolled: the manual ping-pong already provides one iteration of
  // load/compute overlap; full unroll explodes live ranges -> scratch spills
#pragma unroll 1
  for (int k0 = 0; k0 < D_MODEL; k0 += 64) {
    // prefetch k0+32 into set1 (always valid: last k0=704 -> 736 < 768)
    loadA(a1[0], arow0, k0 + 32, lanehi);
    loadA(a1[1], arow1, k0 + 32, lanehi);
#pragma unroll
    for (int t = 0; t < 4; ++t) loadB(b1[t], brow + (size_t)t * 16 * D_MODEL, k0 + 32, lanehi);
    // compute set0 @ k0
#pragma unroll
    for (int i = 0; i < 2; ++i)
#pragma unroll
      for (int t = 0; t < 4; ++t) acc[i][t] = wmma_bf16(a0[i], b0[t], acc[i][t]);
    // prefetch k0+64 into set0
    if (k0 + 64 < D_MODEL) {
      loadA(a0[0], arow0, k0 + 64, lanehi);
      loadA(a0[1], arow1, k0 + 64, lanehi);
#pragma unroll
      for (int t = 0; t < 4; ++t) loadB(b0[t], brow + (size_t)t * 16 * D_MODEL, k0 + 64, lanehi);
    }
    // compute set1 @ k0+32
#pragma unroll
    for (int i = 0; i < 2; ++i)
#pragma unroll
      for (int t = 0; t < 4; ++t) acc[i][t] = wmma_bf16(a1[i], b1[t], acc[i][t]);
  }

#pragma unroll
  for (int i = 0; i < 2; ++i)
#pragma unroll
    for (int r = 0; r < 8; ++r) {
      const int e = e0 + i * 16 + r + 8 * lanehi;    // C layout: M = r + 8*(lane>=16)
      const float bias = b_in[e];
#pragma unroll
      for (int t = 0; t < 4; ++t) {
        const int l = l0 + t * 16 + lanelo;          // N = lane&15
        u_buf[((size_t)b * INNER + e) * SEQLEN + l] = acc[i][t][r] + bias;
      }
    }
}

// ---------------- short filter + split/transpose into bf16 q/k/v ----------------
__global__ void __launch_bounds__(256)
k_filter(const float* __restrict__ u_buf,
         const float* __restrict__ w_sf,
         const float* __restrict__ b_sf,
         uint16_t* __restrict__ x0t,   // [B*H, L, 16]  (q^T)
         uint16_t* __restrict__ x1t,   // [B*H, L, 16]  (k^T)
         uint16_t* __restrict__ vbuf)  // [B*H, 16, L]  (v)
{
  const size_t total = (size_t)BATCH * INNER * SEQLEN;
  size_t idx = (size_t)blockIdx.x * blockDim.x + threadIdx.x;
  if (idx >= total) return;
  const int l = (int)(idx & (SEQLEN - 1));
  const size_t row = idx >> 11;
  const int e = (int)(row % INNER);
  const int b = (int)(row / INNER);

  const float u0  = u_buf[idx];
  const float um1 = (l >= 1) ? u_buf[idx - 1] : 0.f;
  const float um2 = (l >= 2) ? u_buf[idx - 2] : 0.f;
  const float val = w_sf[e*3+0]*um2 + w_sf[e*3+1]*um1 + w_sf[e*3+2]*u0 + b_sf[e];
  const uint16_t bv = f2bf(val);

  const int n = e / 48, c = e % 48;
  const size_t head = (size_t)b * NUM_HEADS + n;
  if (c < 16)       x0t[(head * SEQLEN + l) * 16 + c] = bv;
  else if (c < 32)  x1t[(head * SEQLEN + l) * 16 + (c - 16)] = bv;
  else              vbuf[(head * 16 + (c - 32)) * SEQLEN + l] = bv;
}

// ---------------- kernel-weighted causal attention ----------------
// y[b,l,n,j] = sum_{m<=l} h[n,l-m] * (q_l . k_m) * v[j,m]
// block = 128 thr (4 waves), wave: one 16-row l tile. grid (L/64, 48, B)
__global__ void __launch_bounds__(128)
k_attn(const uint16_t* __restrict__ x0t,
       const uint16_t* __restrict__ x1t,
       const uint16_t* __restrict__ vbuf,
       const float*    __restrict__ h,
       float*          __restrict__ y_attn) {
  __shared__ __align__(16) float    hshm[SEQLEN];
  __shared__ __align__(16) uint16_t sshm[4][16 * 32];

  const int lane = threadIdx.x & 31, wave = threadIdx.x >> 5;
  const int lanehi = lane >> 4, lanelo = lane & 15;
  const int n = blockIdx.y, b = blockIdx.z;
  const size_t head = (size_t)b * NUM_HEADS + n;
  const int l0 = blockIdx.x * 64 + wave * 16;

  for (int i = threadIdx.x; i < SEQLEN; i += blockDim.x)
    hshm[i] = h[(size_t)n * SEQLEN + i];
  __syncthreads();

  // A operand for QK: A[l_local][i] = x0^T; K=16..31 zero-padded
  Frag aq;
  {
    const uint16_t* p = x0t + (head * SEQLEN + l0 + lanelo) * 16;
    aq.q[0] = lanehi ? make_uint4(0u, 0u, 0u, 0u) : *reinterpret_cast<const uint4*>(p);
    aq.q[1] = lanehi ? make_uint4(0u, 0u, 0u, 0u) : *reinterpret_cast<const uint4*>(p + 8);
  }

  uint16_t* stile = &sshm[wave][0];
  const uint16_t* khead = x1t + head * (size_t)SEQLEN * 16;
  const uint16_t* vrow  = vbuf + head * (size_t)16 * SEQLEN + (size_t)lanelo * SEQLEN;

  const v8f z8 = {0.f,0.f,0.f,0.f,0.f,0.f,0.f,0.f};
  v8f y = z8;

  const int nSteps = (l0 + 47) >> 5;
#pragma unroll 1
  for (int s = 0; s < nSteps; ++s) {
    const int m0 = s << 5;

    // B operands for QK: B[i][m_local] -> full 32B row of x1t; lanes>=16 carry K=16..31 (zero)
    Frag bk0, bk1;
    {
      const uint16_t* p0 = khead + (size_t)(m0 + lanelo) * 16;
      bk0.q[0] = *reinterpret_cast<const uint4*>(p0);
      bk0.q[1] = *reinterpret_cast<const uint4*>(p0 + 8);
      const uint16_t* p1 = p0 + 16 * 16;
      bk1.q[0] = *reinterpret_cast<const uint4*>(p1);
      bk1.q[1] = *reinterpret_cast<const uint4*>(p1 + 8);
      if (lanehi) {
        const uint4 z = make_uint4(0u, 0u, 0u, 0u);
        bk0.q[0] = z; bk0.q[1] = z; bk1.q[0] = z; bk1.q[1] = z;
      }
    }
    // issue V load early: only consumed by the final wmma of this step,
    // stays in flight across QK wmmas + h-scaling + LDS round-trip
    Frag bv;
    {
      const uint16_t* p = vrow + m0 + lanehi * 8;
      bv.q[0] = *reinterpret_cast<const uint4*>(p);
      bv.q[1] = *reinterpret_cast<const uint4*>(p + 16);
    }

    v8f s0 = wmma_bf16(aq, bk0, z8);
    v8f s1 = wmma_bf16(aq, bk1, z8);

    // scale by h[l-m] (clamped: negative offset == acausal -> 0), pack bf16, stage in LDS
    const int dl = l0 + 8 * lanehi - m0 - lanelo;   // offset for r=0, first m-subtile
#pragma unroll
    for (int r = 0; r < 8; ++r) {
      int i0 = dl + r;
      int i1 = i0 - 16;
      float h0 = hshm[i0 < 0 ? 0 : i0] * (i0 < 0 ? 0.f : 1.f);
      float h1 = hshm[i1 < 0 ? 0 : i1] * (i1 < 0 ? 0.f : 1.f);
      const int M = r + 8 * lanehi;
      stile[M * 32 + lanelo]      = f2bf(s0[r] * h0);
      stile[M * 32 + 16 + lanelo] = f2bf(s1[r] * h1);
    }
    asm volatile("s_wait_dscnt 0" ::: "memory");

    // reload S as A operand (16x32), wave-private so no barrier needed
    Frag as;
    {
      const uint16_t* p = stile + lanelo * 32 + lanehi * 8;
      as.q[0] = *reinterpret_cast<const uint4*>(p);
      as.q[1] = *reinterpret_cast<const uint4*>(p + 16);
    }
    y = wmma_bf16(as, bv, y);
  }

#pragma unroll
  for (int r = 0; r < 8; ++r) {
    const int M = r + 8 * lanehi;
    y_attn[((size_t)b * SEQLEN + l0 + M) * D_MODEL + n * 16 + lanelo] = y[r];
  }
}

// ---------------- RMS norm (per row of 768) -> bf16 ----------------
__global__ void __launch_bounds__(256)
k_rmsnorm(const float* __restrict__ y_attn,
          const float* __restrict__ norm_w,
          uint16_t*    __restrict__ ynorm) {
  __shared__ float red[256];
  const int row = blockIdx.x;        // b*L + l
  const int tid = threadIdx.x;
  const float* yr = y_attn + (size_t)row * D_MODEL;
  const float v0 = yr[tid], v1 = yr[tid + 256], v2 = yr[tid + 512];
  red[tid] = v0 * v0 + v1 * v1 + v2 * v2;
  __syncthreads();
  for (int off = 128; off > 0; off >>= 1) {
    if (tid < off) red[tid] += red[tid + off];
    __syncthreads();
  }
  const float rs = rsqrtf(red[0] * (1.0f / D_MODEL) + EPSV);
  uint16_t* outr = ynorm + (size_t)row * D_MODEL;
  outr[tid]       = f2bf(v0 * rs * norm_w[tid]);
  outr[tid + 256] = f2bf(v1 * rs * norm_w[tid + 256]);
  outr[tid + 512] = f2bf(v2 * rs * norm_w[tid + 512]);
}

// ---------------- out-projection GEMM + bias + skip ----------------
// block = 256 thr (8 waves). wave: 32l x 64e, rolled ping-pong K pipeline. grid (3, 32, B)
__global__ void __launch_bounds__(256)
k_out_gemm(const uint16_t* __restrict__ ynorm,
           const uint16_t* __restrict__ w_out_bf,
           const float*    __restrict__ b_out,
           const float*    __restrict__ inputs,
           float*          __restrict__ out) {
  const int lane = threadIdx.x & 31, wave = threadIdx.x >> 5;
  const int lanehi = lane >> 4, lanelo = lane & 15;
  const int b  = blockIdx.z;
  const int l0 = blockIdx.y * 64 + (wave & 1) * 32;
  const int e0 = blockIdx.x * 256 + (wave >> 1) * 64;

  const uint16_t* arow0 = ynorm + ((size_t)b * SEQLEN + l0 + lanelo) * D_MODEL;
  const uint16_t* arow1 = arow0 + (size_t)16 * D_MODEL;
  const uint16_t* brow  = w_out_bf + (size_t)(e0 + lanelo) * D_MODEL;

  v8f acc[2][4];
  const v8f z8 = {0.f,0.f,0.f,0.f,0.f,0.f,0.f,0.f};
#pragma unroll
  for (int i = 0; i < 2; ++i)
#pragma unroll
    for (int t = 0; t < 4; ++t) acc[i][t] = z8;

  Frag a0[2], b0[4], a1[2], b1[4];
  loadA(a0[0], arow0, 0, lanehi);
  loadA(a0[1], arow1, 0, lanehi);
#pragma unroll
  for (int t = 0; t < 4; ++t) loadB(b0[t], brow + (size_t)t * 16 * D_MODEL, 0, lanehi);

#pragma unroll 1
  for (int k0 = 0; k0 < D_MODEL; k0 += 64) {
    loadA(a1[0], arow0, k0 + 32, lanehi);
    loadA(a1[1], arow1, k0 + 32, lanehi);
#pragma unroll
    for (int t = 0; t < 4; ++t) loadB(b1[t], brow + (size_t)t * 16 * D_MODEL, k0 + 32, lanehi);
#pragma unroll
    for (int i = 0; i < 2; ++i)
#pragma unroll
      for (int t = 0; t < 4; ++t) acc[i][t] = wmma_bf16(a0[i], b0[t], acc[i][t]);
    if (k0 + 64 < D_MODEL) {
      loadA(a0[0], arow0, k0 + 64, lanehi);
      loadA(a0[1], arow1, k0 + 64, lanehi);
#pragma unroll
      for (int t = 0; t < 4; ++t) loadB(b0[t], brow + (size_t)t * 16 * D_MODEL, k0 + 64, lanehi);
    }
#pragma unroll
    for (int i = 0; i < 2; ++i)
#pragma unroll
      for (int t = 0; t < 4; ++t) acc[i][t] = wmma_bf16(a1[i], b1[t], acc[i][t]);
  }

#pragma unroll
  for (int t = 0; t < 4; ++t) {
    const int e = e0 + t * 16 + lanelo;            // N
    const float bias = b_out[e];
#pragma unroll
    for (int i = 0; i < 2; ++i)
#pragma unroll
      for (int r = 0; r < 8; ++r) {
        const int l = l0 + i * 16 + r + 8 * lanehi; // M
        const size_t idx = ((size_t)b * SEQLEN + l) * D_MODEL + e;
        out[idx] = acc[i][t][r] + bias + inputs[idx] * SKIPV;
      }
  }
}

extern "C" void kernel_launch(void* const* d_in, const int* in_sizes, int n_in,
                              void* d_out, int out_size, void* d_ws, size_t ws_size,
                              hipStream_t stream) {
  (void)in_sizes; (void)n_in; (void)out_size;
  const float* inputs = (const float*)d_in[0];
  const float* w_in   = (const float*)d_in[1];
  const float* b_in   = (const float*)d_in[2];
  const float* w_sf   = (const float*)d_in[3];
  const float* b_sf   = (const float*)d_in[4];
  const float* h      = (const float*)d_in[5];
  const float* norm_w = (const float*)d_in[6];
  const float* w_out  = (const float*)d_in[7];
  const float* b_out  = (const float*)d_in[8];
  float* out = (float*)d_out;

  char* ws = (char*)d_ws;
  // u_buf (37748736 B, f32) dead after filter -> overlay y_attn & ynorm inside it
  float*    u_buf    = (float*)(ws + 0);
  float*    y_attn   = (float*)(ws + 0);                     // 12582912 B
  uint16_t* ynorm    = (uint16_t*)(ws + 12582912);           //  6291456 B (inside u_buf)
  uint16_t* w_in_bf  = (uint16_t*)(ws + 37748736);           //  3538944 B
  uint16_t* in_bf    = (uint16_t*)(ws + 41287680);           //  6291456 B
  uint16_t* x0t      = (uint16_t*)(ws + 47579136);           //  6291456 B
  uint16_t* x1t      = (uint16_t*)(ws + 53870592);           //  6291456 B
  uint16_t* vbuf     = (uint16_t*)(ws + 60162048);           //  6291456 B
  uint16_t* w_out_bf = (uint16_t*)(ws + 66453504);           //  1179648 B -> total 67633152
  if (ws_size < 67633152) return;

  const int n_win  = INNER * D_MODEL;          // 1769472
  const int n_inp  = BATCH * SEQLEN * D_MODEL; // 3145728
  const int n_wout = D_MODEL * D_MODEL;        //  589824
  k_cvt_bf16<<<(n_win  + 255) / 256, 256, 0, stream>>>(w_in,   w_in_bf,  n_win);
  k_cvt_bf16<<<(n_inp  + 255) / 256, 256, 0, stream>>>(inputs, in_bf,    n_inp);
  k_cvt_bf16<<<(n_wout + 255) / 256, 256, 0, stream>>>(w_out,  w_out_bf, n_wout);

  k_in_gemm<<<dim3(8, 36, BATCH), 256, 0, stream>>>(w_in_bf, in_bf, b_in, u_buf);

  const size_t total = (size_t)BATCH * INNER * SEQLEN;       // 9437184
  k_filter<<<(unsigned)((total + 255) / 256), 256, 0, stream>>>(u_buf, w_sf, b_sf, x0t, x1t, vbuf);

  k_attn<<<dim3(SEQLEN / 64, NUM_HEADS, BATCH), 128, 0, stream>>>(x0t, x1t, vbuf, h, y_attn);

  k_rmsnorm<<<BATCH * SEQLEN, 256, 0, stream>>>(y_attn, norm_w, ynorm);

  k_out_gemm<<<dim3(3, 32, BATCH), 256, 0, stream>>>(ynorm, w_out_bf, b_out, inputs, out);
}
